// PixelBlock_83494164234317
// MI455X (gfx1250) — matrix operand order repaired and checked by reading, not compiled, not run on hardware
//
#include <hip/hip_runtime.h>
#include <cstdint>
#include <cstddef>

typedef __attribute__((ext_vector_type(16))) _Float16 v16h;
typedef __attribute__((ext_vector_type(8)))  float    v8f;

#define NTOT  8192   // B*H*W
#define BATCH 8
#define HW    1024

// ---------------------------------------------------------------- transposes
__global__ __launch_bounds__(256) void k_nchw_to_planar(const float* __restrict__ src,
                                                        float* __restrict__ dst, int C) {
  int i = blockIdx.x * 256 + threadIdx.x;          // dst planar index over C*B*HW
  int p = i & (HW - 1);
  int b = (i >> 10) & (BATCH - 1);
  int c = i >> 13;                                 // B*HW = 8192 = 2^13
  dst[i] = src[(((size_t)b * C + c) << 10) + p];
}

__global__ __launch_bounds__(256) void k_planar_to_nchw(const float* __restrict__ src,
                                                        float* __restrict__ dst, int C) {
  int i = blockIdx.x * 256 + threadIdx.x;          // dst NCHW index
  int p = i & (HW - 1);
  int c = (i >> 10) % C;
  int b = i / (C * HW);
  dst[i] = src[(((size_t)c * BATCH + b) << 10) + p];
}

// ---------------------------------------------------------------- elementwise
__global__ __launch_bounds__(256) void k_to_f16(const float* __restrict__ x,
                                                _Float16* __restrict__ y, int doElu) {
  size_t i = (size_t)blockIdx.x * 256 + threadIdx.x;
  float v = x[i];
  if (doElu) v = v > 0.f ? v : (__expf(v) - 1.f);
  y[i] = (_Float16)v;
}

__global__ __launch_bounds__(256) void k_glu_res(const float* __restrict__ gbuf,
                                                 const float* __restrict__ res,
                                                 float* __restrict__ dst, int Cin) {
  size_t i = (size_t)blockIdx.x * 256 + threadIdx.x;   // over Cin*NTOT
  float a  = gbuf[i];
  float bb = gbuf[(size_t)Cin * NTOT + i];
  dst[i] = a * (1.f / (1.f + __expf(-bb))) + res[i];
}

// ---------------------------------------------------------------- weight prep
// Weight-norm w = g*v/||v|| with optional causal zeroing (applied to v BEFORE
// the norm). Output f16 [Cout][Kpad], K ordered as k = (kh*Kw+kw)*Cin + ci,
// zero padded for k >= Cin*Kh*Kw.
__global__ __launch_bounds__(256) void k_prep_w(const float* __restrict__ v,
                                                const float* __restrict__ g,
                                                _Float16* __restrict__ Wout,
                                                int Cin, int Kh, int Kw, int Kpad,
                                                int causal) {
  __shared__ float red[256];
  const int co = blockIdx.x, tid = threadIdx.x;
  const int KK = Kh * Kw, K = Cin * KK;
  const float* vb = v + (size_t)co * K;
  float sum = 0.f;
  for (int k = tid; k < K; k += 256) {
    int ci = k % Cin, pos = k / Cin;
    int kh = pos / Kw, kw = pos - kh * Kw;
    float val = vb[ci * KK + pos];
    if (causal && kh == Kh - 1 && kw >= (Kw >> 1)) val = 0.f;
    sum += val * val;
  }
  red[tid] = sum; __syncthreads();
  for (int s = 128; s > 0; s >>= 1) { if (tid < s) red[tid] += red[tid + s]; __syncthreads(); }
  const float scale = g[co] * rsqrtf(red[0]);
  for (int k = tid; k < Kpad; k += 256) {
    float val = 0.f;
    if (k < K) {
      int ci = k % Cin, pos = k / Cin;
      int kh = pos / Kw, kw = pos - kh * Kw;
      val = vb[ci * KK + pos];
      if (causal && kh == Kh - 1 && kw >= (Kw >> 1)) val = 0.f;
      val *= scale;
    }
    Wout[(size_t)co * Kpad + k] = (_Float16)val;
  }
}

// ---------------------------------------------------------------- GEMM (WMMA)
// Y[m][n] = sum_k W[m][k] * X[k][n]  (+bias, optional +Y, optional f16 store)
// 128x128x32 block tile, 8 waves each owning a 32x64 wave tile (8 WMMAs/step),
// double-buffered LDS with register staging: one __syncthreads per K-step,
// next tile's global loads issued before the WMMAs on the current buffer.
// mode 0: 1x1 (k = ci, guarded by Kin)    mode 1: causal 5x5 implicit im2col,
//   Cin=256, k=(kh*5+kw)*256+ci, pad top=4 left/right=2.
// flags bit0: accumulate into Yf; bit1: write f16 to Yh instead of f32 to Yf.
__global__ __launch_bounds__(256) void k_gemm_wmma(const _Float16* __restrict__ W,
                                                   const _Float16* __restrict__ X,
                                                   const float* __restrict__ bias,
                                                   float* __restrict__ Yf,
                                                   _Float16* __restrict__ Yh,
                                                   int M, int Kpad, int Kin,
                                                   int mode, int flags) {
  __shared__ _Float16 As[2][128][40];     // [buf][m][k]
  __shared__ _Float16 Bs[2][128][40];     // [buf][n][k]
  const int tid  = threadIdx.x;
  const int lane = tid & 31, wave = tid >> 5;
  const int wm   = wave & 3;          // 32-row group of this wave
  const int wn   = wave >> 2;         // 64-col group of this wave
  const int n0   = blockIdx.x * 128;
  const int m0   = blockIdx.y * 128;
  const int khalf = lane >> 4;
  const int mlan  = lane & 15;

  v8f acc[2][4];
  #pragma unroll
  for (int s = 0; s < 2; ++s)
    #pragma unroll
    for (int t = 0; t < 4; ++t)
      acc[s][t] = (v8f){0.f,0.f,0.f,0.f,0.f,0.f,0.f,0.f};

  // register staging for the next tile
  uint4 ast0, ast1;
  alignas(16) _Float16 bst[16];
  const uint4 z4 = {0u, 0u, 0u, 0u};

  const int ar   = tid >> 1;          // A row handled by this thread (0..127)
  const int ac   = (tid & 1) * 16;    // A k-offset (0 or 16)
  const int bkr  = tid >> 3;          // B k-row (0..31)
  const int bnc  = (tid & 7) * 16;    // B n-offset (0..112)

  auto load_tiles = [&](int kb) {
    // A: W[m0..m0+127][kb..kb+31]
    ast0 = z4; ast1 = z4;
    int m = m0 + ar;
    if (m < M) {
      const uint4* p = reinterpret_cast<const uint4*>(W + (size_t)m * Kpad + kb + ac);
      ast0 = p[0]; ast1 = p[1];
      if (kb + 32 < Kpad)
        __builtin_prefetch(W + (size_t)m * Kpad + kb + 32 + ac, 0, 1);
    }
    // B: X gather for rows kb+bkr, cols n0+bnc .. +15
    if (mode == 0) {
      int ci = kb + bkr;
      uint4* bv = reinterpret_cast<uint4*>(bst);
      bv[0] = z4; bv[1] = z4;
      if (ci < Kin) {
        const uint4* p = reinterpret_cast<const uint4*>(X + (size_t)ci * NTOT + n0 + bnc);
        bv[0] = p[0]; bv[1] = p[1];
      }
    } else {
      int k   = kb + bkr;
      int ci  = k & 255;
      int pos = k >> 8;
      int kh  = pos / 5, kw = pos - kh * 5;
      #pragma unroll
      for (int j = 0; j < 16; ++j) {
        int n = n0 + bnc + j;
        int w = n & 31, h = (n >> 5) & 31, b = n >> 10;
        int hh = h + kh - 4, ww = w + kw - 2;
        _Float16 v = (_Float16)0.f;
        if (hh >= 0 && ww >= 0 && ww < 32)  // hh <= h < 32 always
          v = X[(size_t)ci * NTOT + b * 1024 + hh * 32 + ww];
        bst[j] = v;
      }
    }
  };
  auto store_tiles = [&](int buf) {
    *reinterpret_cast<uint4*>(&As[buf][ar][ac])     = ast0;
    *reinterpret_cast<uint4*>(&As[buf][ar][ac + 8]) = ast1;
    #pragma unroll
    for (int j = 0; j < 16; ++j) Bs[buf][bnc + j][bkr] = bst[j];
  };

  const int nsteps = Kpad >> 5;
  load_tiles(0);
  store_tiles(0);

  for (int s = 0; s < nsteps; ++s) {
    __syncthreads();                     // publish buf s&1; retire readers of (s&1)^1
    const int buf = s & 1;
    const bool hasNext = (s + 1) < nsteps;
    if (hasNext) load_tiles((s + 1) << 5);   // global loads in flight during WMMAs

    union F { v16h v; uint4 u[2]; };
    F af[2], bf[4];
    #pragma unroll
    for (int ss = 0; ss < 2; ++ss) {
      const _Float16* ap = &As[buf][wm * 32 + ss * 16 + mlan][0];
      af[ss].u[0] = *reinterpret_cast<const uint4*>(ap + khalf * 8);
      af[ss].u[1] = *reinterpret_cast<const uint4*>(ap + 16 + khalf * 8);
    }
    #pragma unroll
    for (int t = 0; t < 4; ++t) {
      const _Float16* bp = &Bs[buf][wn * 64 + t * 16 + mlan][0];
      bf[t].u[0] = *reinterpret_cast<const uint4*>(bp + khalf * 8);
      bf[t].u[1] = *reinterpret_cast<const uint4*>(bp + 16 + khalf * 8);
    }
    #pragma unroll
    for (int ss = 0; ss < 2; ++ss)
      #pragma unroll
      for (int t = 0; t < 4; ++t)
        acc[ss][t] = __builtin_amdgcn_wmma_f32_16x16x32_f16(
            false, af[ss].v, false, bf[t].v, (short)0, acc[ss][t], false, false);

    if (hasNext) store_tiles(buf ^ 1);
  }

  // ---- epilogue
  #pragma unroll
  for (int ss = 0; ss < 2; ++ss) {
    #pragma unroll
    for (int vv = 0; vv < 8; ++vv) {
      int m = m0 + wm * 32 + ss * 16 + vv + 8 * (lane >> 4);
      if (m < M) {
        float bsm = bias[m];
        #pragma unroll
        for (int t = 0; t < 4; ++t) {
          int n = n0 + wn * 64 + t * 16 + mlan;
          size_t idx = (size_t)m * NTOT + n;
          float r = acc[ss][t][vv] + bsm;
          if (flags & 1) r += Yf[idx];
          if (flags & 2) Yh[idx] = (_Float16)r;
          else           Yf[idx] = r;
        }
      }
    }
  }
}

// ---------------------------------------------------------------- attention
// Flash-style causal attention, dh=16, 8 heads. One block = (batch, head),
// 8 waves x 16 queries = 128 queries per block. grid = (8, B*NH).
__global__ __launch_bounds__(256) void k_attn(const _Float16* __restrict__ Qh,
                                              const _Float16* __restrict__ Kh,
                                              const _Float16* __restrict__ Vh,
                                              float* __restrict__ Out) {
  __shared__ _Float16 Kt[32][16];        // [key][d]
  __shared__ _Float16 Vt[32][16];        // [key][d]
  __shared__ _Float16 Pl[8][16][32];     // per-wave P transpose buffer
  const int tid  = threadIdx.x;
  const int lane = tid & 31, wave = tid >> 5;
  const int nloc = lane & 15, grp = lane >> 4;
  const int bimg = blockIdx.y >> 3, head = blockIdx.y & 7;
  const int qbase = blockIdx.x * 128 + wave * 16;
  const size_t cb = ((size_t)(head * 16)) * NTOT + (size_t)bimg * HW;

  // Q fragment: A matrix 16x32, K dims 16..31 zero-padded
  v16h qf;
  #pragma unroll
  for (int i = 0; i < 16; ++i) {
    int k = ((i >> 3) * 16) + grp * 8 + (i & 7);
    qf[i] = (k < 16) ? Qh[cb + (size_t)k * NTOT + qbase + nloc] : (_Float16)0.f;
  }

  float mrow[8], srow[8];
  v8f oacc = {0.f,0.f,0.f,0.f,0.f,0.f,0.f,0.f};
  #pragma unroll
  for (int vv = 0; vv < 8; ++vv) { mrow[vv] = -3.0e38f; srow[vv] = 0.f; }

  const int pend = blockIdx.x * 128 + 128;
  for (int p0 = 0; p0 < pend; p0 += 32) {
    #pragma unroll
    for (int e = 0; e < 2; ++e) {                // stage 32 keys x 16 d of K and V
      int id2 = tid * 2 + e;
      int key = id2 >> 4, d = id2 & 15;
      Kt[key][d] = Kh[cb + (size_t)d * NTOT + p0 + key];
      Vt[key][d] = Vh[cb + (size_t)d * NTOT + p0 + key];
    }
    __syncthreads();
    if (p0 < qbase + 16) {                       // wave-uniform
      v16h bk0, bk1;
      #pragma unroll
      for (int i = 0; i < 16; ++i) {             // B = K^T [32d x 16key], d>=16 zero
        int dk = ((i >> 3) * 16) + grp * 8 + (i & 7);
        bk0[i] = (dk < 16) ? Kt[nloc][dk]      : (_Float16)0.f;
        bk1[i] = (dk < 16) ? Kt[16 + nloc][dk] : (_Float16)0.f;
      }
      v8f z = {0.f,0.f,0.f,0.f,0.f,0.f,0.f,0.f};
      v8f s0 = __builtin_amdgcn_wmma_f32_16x16x32_f16(false, qf, false, bk0,
                                                      (short)0, z, false, false);
      v8f s1 = __builtin_amdgcn_wmma_f32_16x16x32_f16(false, qf, false, bk1,
                                                      (short)0, z, false, false);
      #pragma unroll
      for (int vv = 0; vv < 8; ++vv) {
        int iq = qbase + vv + 8 * grp;
        float a0 = ((p0 + nloc)      < iq) ? s0[vv] * 0.25f : -1.0e30f;
        float a1 = ((p0 + 16 + nloc) < iq) ? s1[vv] * 0.25f : -1.0e30f;
        float tm = fmaxf(a0, a1);
        #pragma unroll
        for (int msk = 1; msk < 16; msk <<= 1) tm = fmaxf(tm, __shfl_xor(tm, msk, 32));
        float nm   = fmaxf(mrow[vv], tm);
        float corr = __expf(mrow[vv] - nm);
        float e0 = __expf(a0 - nm), e1 = __expf(a1 - nm);
        float rs = e0 + e1;
        #pragma unroll
        for (int msk = 1; msk < 16; msk <<= 1) rs += __shfl_xor(rs, msk, 32);
        srow[vv] = srow[vv] * corr + rs;
        mrow[vv] = nm;
        oacc[vv] = oacc[vv] * corr;
        int mq = vv + 8 * grp;
        Pl[wave][mq][nloc]      = (_Float16)e0;
        Pl[wave][mq][16 + nloc] = (_Float16)e1;
      }
      asm volatile("s_wait_dscnt 0" ::: "memory");   // within-wave LDS transpose
      v16h pf, bv;
      #pragma unroll
      for (int i = 0; i < 16; ++i) {
        int kk = ((i >> 3) * 16) + grp * 8 + (i & 7);
        pf[i] = Pl[wave][nloc][kk];                  // P as A-fragment [16q x 32key]
        bv[i] = Vt[kk][nloc];              // V as B-fragment [32key x 16d]
      }
      oacc = __builtin_amdgcn_wmma_f32_16x16x32_f16(false, pf, false, bv,
                                                    (short)0, oacc, false, false);
    }
    __syncthreads();
  }
  #pragma unroll
  for (int vv = 0; vv < 8; ++vv) {
    int iq = qbase + vv + 8 * grp;
    float den = srow[vv];
    float val = (iq > 0 && den > 0.f) ? oacc[vv] / den : 0.f;   // start_mask row 0
    Out[((size_t)(head * 16 + nloc)) * NTOT + (size_t)bimg * HW + iq] = val;
  }
}

// ---------------------------------------------------------------- host
extern "C" void kernel_launch(void* const* d_in, const int* in_sizes, int n_in,
                              void* d_out, int out_size, void* d_ws, size_t ws_size,
                              hipStream_t stream) {
  (void)in_sizes; (void)n_in; (void)out_size; (void)ws_size;
  // JAX pytree (sorted-key) flatten order:
  // 0 background, 1 input,
  // attn: 2 k.b 3 k.g 4 k.v 5 q.b 6 q.g 7 q.v 8 v.b 9 v.g 10 v.v
  // key_rb: 11 c1.b 12 c1.g 13 c1.v 14 c2.b 15 c2.g 16 c2.v
  // out_rb: 17 c1.b 18 c1.g 19 c1.v 20 c2.b 21 c2.g 22 c2.v 23 ca.b 24 ca.g 25 ca.v
  // query_rb: 26 c1.b 27 c1.g 28 c1.v 29 c2.b 30 c2.g 31 c2.v
  // res[r]: 32+6r: c1.b c1.g c1.v c2.b c2.g c2.v
  auto F = [&](int i) { return (const float*)d_in[i]; };
  const float* bg_in = F(0);
  const float* x_in  = F(1);

  char* w = (char*)d_ws;
  auto carve = [&](size_t bytes) {
    char* p = w; w += (bytes + 255) & ~(size_t)255; return p;
  };
  float* P_IN   = (float*)carve((size_t)256 * NTOT * 4);
  float* P_BG   = (float*)carve((size_t)2   * NTOT * 4);
  float* P_OUT  = (float*)carve((size_t)256 * NTOT * 4);
  float* KCAT   = (float*)carve((size_t)514 * NTOT * 4);
  float* QCAT   = (float*)carve((size_t)258 * NTOT * 4);
  float* G_MID  = (float*)carve((size_t)256 * NTOT * 4);
  float* G_BIG  = (float*)carve((size_t)1028 * NTOT * 4);
  float* P_ATT  = (float*)carve((size_t)128 * NTOT * 4);
  _Float16* E_A = (_Float16*)carve((size_t)544 * NTOT * 2);
  _Float16* E_B = (_Float16*)carve((size_t)384 * NTOT * 2);
  _Float16* WSB = (_Float16*)carve((size_t)512 * 6400 * 2);

  auto prep = [&](const float* v, const float* g, int Cout, int Cin,
                  int Kh, int Kw, int Kpad, int causal) {
    k_prep_w<<<Cout, 256, 0, stream>>>(v, g, WSB, Cin, Kh, Kw, Kpad, causal);
  };
  auto gemm = [&](const _Float16* X, const float* bias, float* Yf, _Float16* Yh,
                  int M, int Kpad, int Kin, int mode, int flags) {
    dim3 grid(NTOT / 128, (M + 127) / 128);
    k_gemm_wmma<<<grid, 256, 0, stream>>>(WSB, X, bias, Yf, Yh, M, Kpad, Kin, mode, flags);
  };
  auto cvt = [&](const float* x, _Float16* y, int planes, int doElu) {
    int n = planes * NTOT;
    k_to_f16<<<n / 256, 256, 0, stream>>>(x, y, doElu);
  };
  auto glu = [&](const float* gbuf, const float* res, float* dst, int Cin) {
    int n = Cin * NTOT;
    k_glu_res<<<n / 256, 256, 0, stream>>>(gbuf, res, dst, Cin);
  };
  auto cpy = [&](void* dst, const void* src, size_t bytes) {
    hipMemcpyAsync(dst, src, bytes, hipMemcpyDeviceToDevice, stream);
  };

  // ---- layout conversion, out = input
  k_nchw_to_planar<<<(256 * NTOT) / 256, 256, 0, stream>>>(x_in, P_IN, 256);
  k_nchw_to_planar<<<(2 * NTOT) / 256, 256, 0, stream>>>(bg_in, P_BG, 2);
  cpy(P_OUT, P_IN, (size_t)256 * NTOT * 4);

  // ---- 4 causal residual blocks
  for (int r = 0; r < 4; ++r) {
    int base = 32 + 6 * r;
    prep(F(base + 2), F(base + 1), 256, 256, 5, 5, 6400, 1);
    cvt(P_OUT, E_A, 256, 1);
    gemm(E_A, F(base + 0), G_MID, nullptr, 256, 6400, 6400, 1, 0);
    prep(F(base + 5), F(base + 4), 512, 256, 5, 5, 6400, 1);
    cvt(G_MID, E_A, 256, 1);
    gemm(E_A, F(base + 3), G_BIG, nullptr, 512, 6400, 6400, 1, 0);
    glu(G_BIG, P_OUT, P_OUT, 256);
  }

  // ---- key_rb on concat(input, out, background)  (514 ch)
  cpy(KCAT,              P_IN,  (size_t)256 * NTOT * 4);
  cpy(KCAT + 256 * NTOT, P_OUT, (size_t)256 * NTOT * 4);
  cpy(KCAT + 512 * NTOT, P_BG,  (size_t)2   * NTOT * 4);
  prep(F(13), F(12), 256, 514, 1, 1, 544, 0);
  cvt(KCAT, E_A, 514, 1);
  gemm(E_A, F(11), G_MID, nullptr, 256, 544, 514, 0, 0);
  prep(F(16), F(15), 1028, 256, 1, 1, 256, 0);
  cvt(G_MID, E_A, 256, 1);
  gemm(E_A, F(14), G_BIG, nullptr, 1028, 256, 256, 0, 0);
  glu(G_BIG, KCAT, KCAT, 514);                        // KCAT now = key

  // ---- query_rb on concat(out, background)  (258 ch)
  cpy(QCAT,              P_OUT, (size_t)256 * NTOT * 4);
  cpy(QCAT + 256 * NTOT, P_BG,  (size_t)2   * NTOT * 4);
  prep(F(28), F(27), 256, 258, 1, 1, 288, 0);
  cvt(QCAT, E_A, 258, 1);
  gemm(E_A, F(26), G_MID, nullptr, 256, 288, 258, 0, 0);
  prep(F(31), F(30), 516, 256, 1, 1, 256, 0);
  cvt(G_MID, E_A, 256, 1);
  gemm(E_A, F(29), G_BIG, nullptr, 516, 256, 256, 0, 0);
  glu(G_BIG, QCAT, QCAT, 258);                        // QCAT now = query

  // ---- attention linears -> f16 Q/K/V planar [128][NTOT]
  _Float16* Qh = E_B;
  _Float16* Kh = E_B + (size_t)128 * NTOT;
  _Float16* Vh = E_B + (size_t)256 * NTOT;
  cvt(QCAT, E_A, 258, 0);
  prep(F(7), F(6), 128, 258, 1, 1, 288, 0);
  gemm(E_A, F(5), nullptr, Qh, 128, 288, 258, 0, 2);
  cvt(KCAT, E_A, 514, 0);
  prep(F(4), F(3), 128, 514, 1, 1, 544, 0);
  gemm(E_A, F(2), nullptr, Kh, 128, 544, 514, 0, 2);
  prep(F(10), F(9), 128, 514, 1, 1, 544, 0);
  gemm(E_A, F(8), nullptr, Vh, 128, 544, 514, 0, 2);

  // ---- causal attention
  k_attn<<<dim3(8, BATCH * 8), 256, 0, stream>>>(Qh, Kh, Vh, P_ATT);

  // ---- out_rb with aux = attention output
  prep(F(19), F(18), 256, 256, 1, 1, 256, 0);
  cvt(P_OUT, E_A, 256, 1);
  gemm(E_A, F(17), G_MID, nullptr, 256, 256, 256, 0, 0);
  prep(F(25), F(24), 256, 128, 1, 1, 128, 0);
  cvt(P_ATT, E_A, 128, 1);
  gemm(E_A, F(23), G_MID, nullptr, 256, 128, 128, 0, 1);   // accumulate
  prep(F(22), F(21), 512, 256, 1, 1, 256, 0);
  cvt(G_MID, E_A, 256, 1);
  gemm(E_A, F(20), G_BIG, nullptr, 512, 256, 256, 0, 0);
  glu(G_BIG, P_OUT, P_OUT, 256);

  // ---- back to NCHW
  k_planar_to_nchw<<<(256 * NTOT) / 256, 256, 0, stream>>>(P_OUT, (float*)d_out, 256);
}
